// VectorQuantizer_65712999629501
// MI455X (gfx1250) — compile-verified, matrix operand order checked by reading, not run on hardware
//
#include <hip/hip_runtime.h>
#include <hip/hip_bf16.h>

typedef __bf16 bf16;
typedef bf16  v16bf __attribute__((ext_vector_type(16)));
typedef float v8f   __attribute__((ext_vector_type(8)));

#define N_ROWS 16384
#define D_IN   840
#define D_CBV  1024
#define KCODE  1024
#define BETA   0.3f

#define TM 128
#define TN 128
#define TK 32
#define LDT 34  // padded k-stride in LDS (bf16 elems): stride 68B -> conflict-free

static __device__ __forceinline__ v8f wmma_bf16(v16bf a, v16bf b, v8f c) {
  return __builtin_amdgcn_wmma_f32_16x16x32_bf16(false, a, false, b, (short)0, c,
                                                 false, false);
}

// 16-bit A-matrix 16x32 fragment layout (ISA 7.12.2):
// lane<16: V0..3 -> K{0..7}, V4..7 -> K{16..23}; lane>=16: +8.
// B is staged N-major ([n][k]) so the same loader fills B fragments.
static __device__ __forceinline__ v16bf load_frag(const bf16* base) {
  int lane = threadIdx.x & 31;
  const bf16* p = base + (lane & 15) * LDT + ((lane & 16) ? 8 : 0);
  v16bf f;
#pragma unroll
  for (int v = 0; v < 8; ++v) {
    int k = ((v & 4) ? 16 : 0) + ((v & 3) << 1);
    f[2 * v]     = p[k];
    f[2 * v + 1] = p[k + 1];
  }
  return f;
}

static __device__ __forceinline__ void split2(float x, bf16& h, bf16& l) {
  h = (bf16)x;
  l = (bf16)(x - (float)h);
}

static __device__ __forceinline__ unsigned int fkey(float f) {
  unsigned int u = __float_as_uint(f);
  return (u & 0x80000000u) ? ~u : (u | 0x80000000u);
}

// ---------------- GEMM1: z_e = x @ W + b ----------------
// WG tile 128x128, 8 waves; each wave: 32 rows x 64 cols = 2x4 WMMA tiles.
__global__ __launch_bounds__(256) void k_gemm_ze(const float* __restrict__ x,
                                                 const float* __restrict__ W,
                                                 const float* __restrict__ bias,
                                                 float* __restrict__ ze) {
  __shared__ bf16 sAh[TM * LDT], sAl[TM * LDT];
  __shared__ bf16 sBh[TN * LDT], sBl[TN * LDT];
  const int row0 = blockIdx.y * TM;
  const int col0 = blockIdx.x * TN;
  const int t = threadIdx.x;
  const int wid = t >> 5, wr = wid & 3, wc = wid >> 2;

  v8f acc[2][4];
  v8f zv = {};
#pragma unroll
  for (int i = 0; i < 2; ++i)
#pragma unroll
    for (int j = 0; j < 4; ++j) acc[i][j] = zv;

  const int ksteps = (D_IN + TK - 1) / TK;  // 27
  for (int s = 0; s < ksteps; ++s) {
    const int k0 = s * TK;
#pragma unroll
    for (int i = 0; i < (TM * TK) / 256; ++i) {  // A: coalesced over k
      int e = t + i * 256;
      int r = e >> 5, kk = e & 31, gk = k0 + kk;
      float v = (gk < D_IN) ? x[(size_t)(row0 + r) * D_IN + gk] : 0.0f;
      bf16 h, l; split2(v, h, l);
      sAh[r * LDT + kk] = h; sAl[r * LDT + kk] = l;
    }
#pragma unroll
    for (int i = 0; i < (TN * TK) / 256; ++i) {  // B: coalesced over n
      int e = t + i * 256;
      int n = e & 127, kk = e >> 7, gk = k0 + kk;
      float v = (gk < D_IN) ? W[(size_t)gk * D_CBV + col0 + n] : 0.0f;
      bf16 h, l; split2(v, h, l);
      sBh[n * LDT + kk] = h; sBl[n * LDT + kk] = l;
    }
    __syncthreads();

    v16bf ah[2], al[2], bh[4], bl[4];
#pragma unroll
    for (int rt = 0; rt < 2; ++rt) {
      ah[rt] = load_frag(sAh + (wr * 32 + rt * 16) * LDT);
      al[rt] = load_frag(sAl + (wr * 32 + rt * 16) * LDT);
    }
#pragma unroll
    for (int ct = 0; ct < 4; ++ct) {
      bh[ct] = load_frag(sBh + (wc * 64 + ct * 16) * LDT);
      bl[ct] = load_frag(sBl + (wc * 64 + ct * 16) * LDT);
    }
#pragma unroll
    for (int rt = 0; rt < 2; ++rt)
#pragma unroll
      for (int ct = 0; ct < 4; ++ct) {
        acc[rt][ct] = wmma_bf16(al[rt], bh[ct], acc[rt][ct]);
        acc[rt][ct] = wmma_bf16(ah[rt], bl[ct], acc[rt][ct]);
        acc[rt][ct] = wmma_bf16(ah[rt], bh[ct], acc[rt][ct]);
      }
    __syncthreads();
  }

  const int lane = t & 31, ln = lane & 15, lr = (lane & 16) ? 8 : 0;
#pragma unroll
  for (int rt = 0; rt < 2; ++rt)
#pragma unroll
    for (int ct = 0; ct < 4; ++ct) {
      int gcol = col0 + wc * 64 + ct * 16 + ln;
      float bv = bias[gcol];
#pragma unroll
      for (int v = 0; v < 8; ++v) {
        int grow = row0 + wr * 32 + rt * 16 + lr + v;
        ze[(size_t)grow * D_CBV + gcol] = acc[rt][ct][v] + bv;
      }
    }
}

// ---------------- GEMM2: fused distance + argmin ----------------
__global__ __launch_bounds__(256) void k_gemm_dist(
    const float* __restrict__ ze, const float* __restrict__ cb,
    const float* __restrict__ znorm, const float* __restrict__ cnorm,
    unsigned long long* __restrict__ best) {
  __shared__ bf16 sAh[TM * LDT], sAl[TM * LDT];
  __shared__ bf16 sBh[TN * LDT], sBl[TN * LDT];
  __shared__ unsigned long long cand[TM];
  const int row0 = blockIdx.y * TM;
  const int col0 = blockIdx.x * TN;
  const int t = threadIdx.x;
  const int wid = t >> 5, wr = wid & 3, wc = wid >> 2;
  if (t < TM) cand[t] = ~0ull;

  v8f acc[2][4];
  v8f zv = {};
#pragma unroll
  for (int i = 0; i < 2; ++i)
#pragma unroll
    for (int j = 0; j < 4; ++j) acc[i][j] = zv;

  for (int s = 0; s < D_CBV / TK; ++s) {  // 32 steps
    const int k0 = s * TK;
#pragma unroll
    for (int i = 0; i < (TM * TK) / 256; ++i) {
      int e = t + i * 256;
      int r = e >> 5, kk = e & 31;
      float v = ze[(size_t)(row0 + r) * D_CBV + k0 + kk];
      bf16 h, l; split2(v, h, l);
      sAh[r * LDT + kk] = h; sAl[r * LDT + kk] = l;
    }
#pragma unroll
    for (int i = 0; i < (TN * TK) / 256; ++i) {  // codebook rows are N-major already
      int e = t + i * 256;
      int kk = e & 31, n = e >> 5;
      float v = cb[(size_t)(col0 + n) * D_CBV + k0 + kk];
      bf16 h, l; split2(v, h, l);
      sBh[n * LDT + kk] = h; sBl[n * LDT + kk] = l;
    }
    __syncthreads();

    v16bf ah[2], al[2], bh[4], bl[4];
#pragma unroll
    for (int rt = 0; rt < 2; ++rt) {
      ah[rt] = load_frag(sAh + (wr * 32 + rt * 16) * LDT);
      al[rt] = load_frag(sAl + (wr * 32 + rt * 16) * LDT);
    }
#pragma unroll
    for (int ct = 0; ct < 4; ++ct) {
      bh[ct] = load_frag(sBh + (wc * 64 + ct * 16) * LDT);
      bl[ct] = load_frag(sBl + (wc * 64 + ct * 16) * LDT);
    }
#pragma unroll
    for (int rt = 0; rt < 2; ++rt)
#pragma unroll
      for (int ct = 0; ct < 4; ++ct) {
        acc[rt][ct] = wmma_bf16(al[rt], bh[ct], acc[rt][ct]);
        acc[rt][ct] = wmma_bf16(ah[rt], bl[ct], acc[rt][ct]);
        acc[rt][ct] = wmma_bf16(ah[rt], bh[ct], acc[rt][ct]);
      }
    __syncthreads();
  }

  const int lane = t & 31, ln = lane & 15, lr = (lane & 16) ? 8 : 0;
#pragma unroll
  for (int rt = 0; rt < 2; ++rt)
#pragma unroll
    for (int ct = 0; ct < 4; ++ct) {
      int gcol = col0 + wc * 64 + ct * 16 + ln;
      float cn = cnorm[gcol];
#pragma unroll
      for (int v = 0; v < 8; ++v) {
        int lrow = wr * 32 + rt * 16 + lr + v;
        float d = znorm[row0 + lrow] + cn - 2.0f * acc[rt][ct][v];
        unsigned long long pk =
            ((unsigned long long)fkey(d) << 32) | (unsigned int)gcol;
        atomicMin(&cand[lrow], pk);
      }
    }
  __syncthreads();
  if (t < TM) atomicMin(&best[row0 + t], cand[t]);
}

// ---------------- small kernels ----------------
__global__ void k_init(unsigned long long* best, int* counts, float* loss) {
  int i = blockIdx.x * blockDim.x + threadIdx.x;
  if (i < N_ROWS) best[i] = ~0ull;
  if (i < KCODE) counts[i] = 0;
  if (i == 0) *loss = 0.0f;
}

__global__ void k_zero(float* p, int n) {
  int i = blockIdx.x * blockDim.x + threadIdx.x;
  if (i < n) p[i] = 0.0f;
}

__global__ void k_rownorm(const float* __restrict__ in, float* __restrict__ out,
                          int cols) {
  int row = blockIdx.x * 8 + (threadIdx.x >> 5);
  int lane = threadIdx.x & 31;
  const float* p = in + (size_t)row * cols;
  float s = 0.0f;
  for (int c = lane; c < cols; c += 32) { float v = p[c]; s += v * v; }
#pragma unroll
  for (int off = 16; off > 0; off >>= 1) s += __shfl_xor(s, off, 32);
  if (lane == 0) out[row] = s;
}

__global__ void k_assign(const unsigned long long* __restrict__ best,
                         int* __restrict__ minidx, int* __restrict__ counts,
                         float* __restrict__ enc) {
  int n = blockIdx.x * blockDim.x + threadIdx.x;
  if (n < N_ROWS) {
    int idx = (int)(best[n] & 0xFFFFFFFFu);
    minidx[n] = idx;
    atomicAdd(&counts[idx], 1);
    enc[(size_t)n * KCODE + idx] = 1.0f;
  }
}

__global__ void k_quant_loss(const float* __restrict__ ze,
                             const float* __restrict__ cb,
                             const int* __restrict__ minidx,
                             float* __restrict__ qout, float* __restrict__ loss) {
  int n = blockIdx.x, t = threadIdx.x;
  int idx = minidx[n];
  const float* c = cb + (size_t)idx * D_CBV;
  const float* z = ze + (size_t)n * D_CBV;
  float s = 0.0f;
  for (int j = t; j < D_CBV; j += 256) {
    float q = c[j];
    float d = q - z[j];
    qout[(size_t)n * D_CBV + j] = q;
    s += d * d;
  }
  __shared__ float red[8];
#pragma unroll
  for (int off = 16; off > 0; off >>= 1) s += __shfl_xor(s, off, 32);
  if ((t & 31) == 0) red[t >> 5] = s;
  __syncthreads();
  if (t == 0) {
    float v = 0.0f;
#pragma unroll
    for (int i = 0; i < 8; ++i) v += red[i];
    atomicAdd(loss, v);
  }
}

__global__ void k_scalars(const int* __restrict__ counts,
                          const float* __restrict__ loss, float* out_vq,
                          float* out_perp) {
  int t = threadIdx.x;
  float s = 0.0f;
  const float invN = 1.0f / (float)N_ROWS;
  for (int k = t; k < KCODE; k += 256) {
    float p = (float)counts[k] * invN;
    s += p * logf(p + 1e-10f);
  }
  __shared__ float red[8];
#pragma unroll
  for (int off = 16; off > 0; off >>= 1) s += __shfl_xor(s, off, 32);
  if ((t & 31) == 0) red[t >> 5] = s;
  __syncthreads();
  if (t == 0) {
    float h = 0.0f;
#pragma unroll
    for (int i = 0; i < 8; ++i) h += red[i];
    *out_perp = expf(-h);
    *out_vq = (*loss) * ((1.0f + BETA) / ((float)N_ROWS * (float)D_CBV));
  }
}

extern "C" void kernel_launch(void* const* d_in, const int* in_sizes, int n_in,
                              void* d_out, int out_size, void* d_ws, size_t ws_size,
                              hipStream_t stream) {
  const float* x  = (const float*)d_in[0];
  const float* W  = (const float*)d_in[1];
  const float* b  = (const float*)d_in[2];
  const float* cb = (const float*)d_in[3];

  float* out = (float*)d_out;
  float* out_vq   = out;                                   // scalar
  float* out_q    = out + 1;                               // [N, D_CB]
  float* out_perp = out + 1 + (size_t)N_ROWS * D_CBV;      // scalar
  float* out_enc  = out + 2 + (size_t)N_ROWS * D_CBV;      // [N, K]

  char* w = (char*)d_ws;
  float* ze = (float*)w;                     w += (size_t)N_ROWS * D_CBV * 4;
  float* znorm = (float*)w;                  w += (size_t)N_ROWS * 4;
  float* cnorm = (float*)w;                  w += (size_t)KCODE * 4;
  unsigned long long* best = (unsigned long long*)w; w += (size_t)N_ROWS * 8;
  int* minidx = (int*)w;                     w += (size_t)N_ROWS * 4;
  int* counts = (int*)w;                     w += (size_t)KCODE * 4;
  float* loss = (float*)w;

  k_init<<<(N_ROWS + 255) / 256, 256, 0, stream>>>(best, counts, loss);

  const int enc_n = N_ROWS * KCODE;
  k_zero<<<(enc_n + 255) / 256, 256, 0, stream>>>(out_enc, enc_n);

  k_gemm_ze<<<dim3(D_CBV / TN, N_ROWS / TM), 256, 0, stream>>>(x, W, b, ze);

  k_rownorm<<<N_ROWS / 8, 256, 0, stream>>>(ze, znorm, D_CBV);
  k_rownorm<<<KCODE / 8, 256, 0, stream>>>(cb, cnorm, D_CBV);

  k_gemm_dist<<<dim3(KCODE / TN, N_ROWS / TM), 256, 0, stream>>>(ze, cb, znorm,
                                                                 cnorm, best);

  k_assign<<<N_ROWS / 256, 256, 0, stream>>>(best, minidx, counts, out_enc);

  k_quant_loss<<<N_ROWS, 256, 0, stream>>>(ze, cb, minidx, out_q, loss);

  k_scalars<<<1, 256, 0, stream>>>(counts, loss, out_vq, out_perp);
}
